// MNET_Jelly_70892730187937
// MI455X (gfx1250) — matrix operand order repaired and checked by reading, not compiled, not run on hardware
//
#include <hip/hip_runtime.h>
#include <hip/hip_bf16.h>
#include <math.h>

// ---------------------------------------------------------------------------
// CDNA5 (gfx1250) spiking CNN forward.
// Convs + FCs via implicit-GEMM using v_wmma_f32_16x16x32_bf16 (wave32),
// separable im2col addressing (rowBase(m) + koff(k)), branchless clamped
// gathers, async global->LDS B-tile copies (global_load_async_to_lds_b128 +
// s_wait_asynccnt) with a plain uint4 fallback.
// BN(train-mode) stats via block reductions + f32 atomics, then fused
// BN-normalize + PLIF(T=5, hard reset) + 2x2 maxpool emitting bf16 spikes.
// ---------------------------------------------------------------------------

typedef __attribute__((ext_vector_type(16))) __bf16 v16bf;
typedef __attribute__((ext_vector_type(8)))  float  v8f;
typedef int v4i_g __attribute__((vector_size(16)));  // matches async-LDS builtin pointee

#define TPB 256

#if defined(__AMDGCN__) && __has_builtin(__builtin_amdgcn_global_load_async_to_lds_b128) && __has_builtin(__builtin_amdgcn_s_wait_asynccnt)
#define HAVE_ASYNC_LDS 1
#else
#define HAVE_ASYNC_LDS 0
#endif

__device__ __forceinline__ unsigned short f2bf(float f) {
  unsigned int u = __float_as_uint(f);
  unsigned int r = u + 0x7FFFu + ((u >> 16) & 1u);  // round-to-nearest-even
  return (unsigned short)(r >> 16);
}

// ---------------- conversion kernels ----------------
__global__ void k_f32_to_bf16(const float* __restrict__ in,
                              unsigned short* __restrict__ out, int n) {
  int i = blockIdx.x * blockDim.x + threadIdx.x;
  if (i < n) out[i] = f2bf(in[i]);
}

// conv weights [Cout][Cin][5][5] f32 -> Bmat [Cout][Kpad] bf16 (n-major, 0-pad)
__global__ void k_convw_to_mat(const float* __restrict__ w,
                               unsigned short* __restrict__ out,
                               int Cout, int Cin, int Kpad) {
  int i = blockIdx.x * blockDim.x + threadIdx.x;
  int total = Cout * Kpad;
  if (i >= total) return;
  int n = i / Kpad, k = i - n * Kpad;
  int K = Cin * 25;
  unsigned short v = 0;
  if (k < K) {
    int ci = k / 25, r = k - ci * 25;
    v = f2bf(w[(n * Cin + ci) * 25 + r]);
  }
  out[i] = v;
}

// fc weights [Nreal][K] f32 -> Bmat [Npad][K] bf16 (already n-major, 0-pad rows)
__global__ void k_fcw_to_mat(const float* __restrict__ w,
                             unsigned short* __restrict__ out,
                             int Nreal, int Npad, int K) {
  int i = blockIdx.x * blockDim.x + threadIdx.x;
  int total = Npad * K;
  if (i >= total) return;
  int n = i / K;
  out[i] = (n < Nreal) ? f2bf(w[i]) : (unsigned short)0;
}

__global__ void k_zero_f32(float* p, int n) {
  int i = blockIdx.x * blockDim.x + threadIdx.x;
  if (i < n) p[i] = 0.f;
}

// ---------------- WMMA GEMM (implicit-conv or plain A) ----------------
// A (bf16): CONV=1 -> feature map [Nimg][Cin][Hin][Win]; A[m,k] separable as
//           rowBase(m)+koff(k).  CONV=0 -> plain row-major [M][K].
// Bm (bf16): n-major [Npad][Kpad], zero filled padding.
// Out (f32): CONV=1 -> [img][n][Hc*Wc]; CONV=0 -> [m][Nout].
// 8 waves; each wave owns 16 M-rows x N_TILE cols -> NW WMMAs per k-step
// reusing one A fragment.
template<int N_TILE, int CONV>
__global__ __launch_bounds__(TPB)
void k_gemm_wmma(const unsigned short* __restrict__ A,
                 const unsigned short* __restrict__ Bm,
                 float* __restrict__ Out,
                 int M_total, int K, int Nout,
                 int Cin, int Hin, int Win, int Hc, int Wc) {
  constexpr int NW     = N_TILE / 16;
  constexpr int M_TILE = 128;  // 8 waves * 16 rows

  __shared__ unsigned short As[M_TILE][32];
  __shared__ unsigned short Bs[N_TILE][32];

  const int tid   = threadIdx.x;
  const int mBase = blockIdx.x * M_TILE;
  const int nBase = blockIdx.y * N_TILE;
  const int Kpad  = (K + 31) & ~31;
  const int HWc   = Hc * Wc;

  const int wave = tid >> 5, lane = tid & 31;
  const int subM = wave * 16;
  const int half = lane >> 4, l15 = lane & 15;

  // ---- per-thread gather plan: fixed column kk, 16 fixed rows mr0+8j ----
  const int kk  = tid & 31;
  const int mr0 = tid >> 5;
  int rb[16];
#pragma unroll
  for (int j = 0; j < 16; ++j) {
    int m = mBase + mr0 + j * 8;
    int r = -1;
    if (m < M_total) {
      if (CONV) {
        int img = m / HWc, p = m - img * HWc;
        int oh = p / Wc,  ow = p - oh * Wc;
        r = (img * Cin * Hin + oh) * Win + ow;
      } else {
        r = m * K;
      }
    }
    rb[j] = r;
  }

  v8f acc[NW];
#pragma unroll
  for (int nw = 0; nw < NW; ++nw) acc[nw] = (v8f){};

  for (int k0 = 0; k0 < Kpad; k0 += 32) {
    // ---- A tile: 16 branchless clamped gathers, one koff decompose ----
    int k = k0 + kk;
    bool kvalid = (k < K);
    int ko;
    if (CONV) {
      int ci = k / 25, rr = k - ci * 25;
      int kh = rr / 5, kw = rr - kh * 5;
      ko = (ci * Hin + kh) * Win + kw;
    } else {
      ko = k;
    }
#pragma unroll
    for (int j = 0; j < 16; ++j) {
      bool ok = kvalid && (rb[j] >= 0);
      int idx = ok ? (rb[j] + ko) : 0;       // clamp: offset 0 always readable
      unsigned short t = A[(size_t)idx];     // unconditional load, no exec dance
      As[mr0 + j * 8][kk] = ok ? t : (unsigned short)0;
    }
    // ---- B tile: N_TILE rows of 64B from n-major Bm ----
#if HAVE_ASYNC_LDS
    if (tid < N_TILE * 4) {
      int nn = tid >> 2, ch = tid & 3;
      const unsigned short* gp = Bm + (size_t)(nBase + nn) * Kpad + k0 + ch * 8;
      __builtin_amdgcn_global_load_async_to_lds_b128(
          (__attribute__((address_space(1))) v4i_g*)gp,
          (__attribute__((address_space(3))) v4i_g*)&Bs[nn][ch * 8],
          0, 0);
      if (k0 + 32 < Kpad) __builtin_prefetch(gp + 32, 0, 1);
    }
    __builtin_amdgcn_s_wait_asynccnt(0);
#else
    if (tid < N_TILE * 4) {
      int nn = tid >> 2, ch = tid & 3;
      const uint4* gp = (const uint4*)(Bm + (size_t)(nBase + nn) * Kpad + k0) + ch;
      *((uint4*)&Bs[nn][0] + ch) = *gp;
      if (k0 + 32 < Kpad) __builtin_prefetch((const unsigned short*)gp + 32, 0, 1);
    }
#endif
    __syncthreads();

    // ---- fragments per CDNA5 16-bit WMMA VGPR layouts ----
    union U { unsigned int u[8]; v16bf v; };
    U af;
    const unsigned short* arow = &As[subM + l15][0];
#pragma unroll
    for (int q = 0; q < 8; ++q) {
      // A 16x32: vgpr q<4 -> K = 2q + half*8 ; q>=4 -> K = 16 + 2(q-4) + half*8
      int ka = ((q < 4) ? (2 * q) : (16 + 2 * (q - 4))) + half * 8;
      af.u[q] = *(const unsigned int*)(arow + ka);
    }
#pragma unroll
    for (int nw = 0; nw < NW; ++nw) {
      U bfb;
      const unsigned short* brow = &Bs[nw * 16 + l15][0];
#pragma unroll
      for (int q = 0; q < 8; ++q)  // B 32x16: vgpr q<4 -> K = 2q + half*8 ; q>=4 -> K = 16 + 2(q-4) + half*8
        bfb.u[q] = *(const unsigned int*)(brow + ((q < 4) ? (2 * q) : (16 + 2 * (q - 4))) + half * 8);
      acc[nw] = __builtin_amdgcn_wmma_f32_16x16x32_bf16(
          false, af.v, false, bfb.v, (short)0, acc[nw], false, false);
    }
    __syncthreads();
  }

  // ---- epilogue: D layout vgpr j -> M = j + half*8, N = lane&15 ----
#pragma unroll
  for (int j = 0; j < 8; ++j) {
    int m = mBase + subM + half * 8 + j;
    if (m < M_total) {
      if (CONV) {
        int img = m / HWc, p = m - img * HWc;  // once per row, not per (nw,row)
        float* orow = Out + (size_t)img * Nout * HWc + p;
#pragma unroll
        for (int nw = 0; nw < NW; ++nw)
          orow[(size_t)(nBase + nw * 16 + l15) * HWc] = acc[nw][j];
      } else {
        float* orow = Out + (size_t)m * Nout + nBase + l15;
#pragma unroll
        for (int nw = 0; nw < NW; ++nw)
          orow[nw * 16] = acc[nw][j];
      }
    }
  }
}

// ---------------- BN statistics (sum, sumsq per channel) ----------------
__global__ void k_bn_stats(const float* __restrict__ Y, float* __restrict__ stats,
                           int C, int S, int Nimg) {
  __shared__ float red[TPB * 2];
  int c = blockIdx.x;
  long total = (long)Nimg * S;
  float s = 0.f, s2 = 0.f;
  for (long i = (long)blockIdx.y * TPB + threadIdx.x; i < total;
       i += (long)gridDim.y * TPB) {
    int img = (int)(i / S);
    int sp  = (int)(i - (long)img * S);
    float v = Y[((size_t)img * C + c) * S + sp];
    s += v; s2 += v * v;
  }
  red[threadIdx.x] = s;
  red[TPB + threadIdx.x] = s2;
  __syncthreads();
  for (int off = TPB / 2; off > 0; off >>= 1) {
    if (threadIdx.x < off) {
      red[threadIdx.x]       += red[threadIdx.x + off];
      red[TPB + threadIdx.x] += red[TPB + threadIdx.x + off];
    }
    __syncthreads();
  }
  if (threadIdx.x == 0) {
    atomicAdd(&stats[2 * c],     red[0]);
    atomicAdd(&stats[2 * c + 1], red[TPB]);
  }
}

// ---------------- fused BN-normalize + PLIF(T=5) + 2x2 maxpool ----------------
// timeInvariant=1: conv output identical across T (layer 1), Y has Nimg=B.
__global__ void k_bn_plif_pool(const float* __restrict__ Y,
                               const float* __restrict__ stats,
                               const float* __restrict__ gamma,
                               const float* __restrict__ beta,
                               const float* __restrict__ wlif,
                               unsigned short* __restrict__ Sout,
                               int Bsz, int C, int Hc, int Wc, int Hp, int Wp,
                               float invCount, int timeInvariant) {
  int i = blockIdx.x * blockDim.x + threadIdx.x;
  int total = Bsz * C * Hp * Wp;
  if (i >= total) return;
  int pw = i % Wp; int t1 = i / Wp;
  int ph = t1 % Hp; int t2 = t1 / Hp;
  int c  = t2 % C;  int b  = t2 / C;

  float mean  = stats[2 * c] * invCount;
  float var   = stats[2 * c + 1] * invCount - mean * mean;
  float scale = gamma[c] * rsqrtf(var + 1e-5f);
  float shift = beta[c] - mean * scale;
  float decay = 1.f / (1.f + __expf(-wlif[0]));

  float v0 = 0.f, v1 = 0.f, v2 = 0.f, v3 = 0.f;
  for (int t = 0; t < 5; ++t) {
    int imgY = timeInvariant ? b : (t * Bsz + b);
    size_t base = (((size_t)imgY * C + c) * Hc + 2 * ph) * Wc + 2 * pw;
    float x0 = Y[base]          * scale + shift;
    float x1 = Y[base + 1]      * scale + shift;
    float x2 = Y[base + Wc]     * scale + shift;
    float x3 = Y[base + Wc + 1] * scale + shift;
    v0 += (x0 - v0) * decay;  v1 += (x1 - v1) * decay;
    v2 += (x2 - v2) * decay;  v3 += (x3 - v3) * decay;
    float s0 = (v0 >= 1.f) ? 1.f : 0.f;
    float s1 = (v1 >= 1.f) ? 1.f : 0.f;
    float s2 = (v2 >= 1.f) ? 1.f : 0.f;
    float s3 = (v3 >= 1.f) ? 1.f : 0.f;
    v0 *= (1.f - s0); v1 *= (1.f - s1); v2 *= (1.f - s2); v3 *= (1.f - s3);
    float sp = fmaxf(fmaxf(s0, s1), fmaxf(s2, s3));
    Sout[(((size_t)(t * Bsz + b) * C + c) * Hp + ph) * Wp + pw] =
        (sp > 0.f) ? (unsigned short)0x3F80 : (unsigned short)0;  // bf16 1.0/0.0
  }
}

// ---------------- FC1 PLIF (bias + scan over T, spike out bf16) ----------------
__global__ void k_fc_plif(const float* __restrict__ Yfc,
                          const float* __restrict__ bias,
                          const float* __restrict__ wlif,
                          unsigned short* __restrict__ Sout,
                          int Bsz, int N) {
  int i = blockIdx.x * blockDim.x + threadIdx.x;
  if (i >= Bsz * N) return;
  int o = i % N, b = i / N;
  float decay = 1.f / (1.f + __expf(-wlif[0]));
  float bo = bias[o];
  float v = 0.f;
  for (int t = 0; t < 5; ++t) {
    float x = Yfc[(size_t)(t * Bsz + b) * N + o] + bo;
    v += (x - v) * decay;
    float s = (v >= 1.f) ? 1.f : 0.f;
    v *= (1.f - s);
    Sout[(size_t)(t * Bsz + b) * N + o] =
        (s > 0.f) ? (unsigned short)0x3F80 : (unsigned short)0;
  }
}

// ---------------- FC2 PLIF + mean over time -> output [B,10] f32 ----------------
__global__ void k_fc2_plif_mean(const float* __restrict__ Yfc,
                                const float* __restrict__ bias,
                                const float* __restrict__ wlif,
                                float* __restrict__ out, int Bsz) {
  int i = blockIdx.x * blockDim.x + threadIdx.x;
  if (i >= Bsz * 10) return;
  int o = i % 10, b = i / 10;
  float decay = 1.f / (1.f + __expf(-wlif[0]));
  float bo = bias[o];
  float v = 0.f, accum = 0.f;
  for (int t = 0; t < 5; ++t) {
    float x = Yfc[(size_t)(t * Bsz + b) * 16 + o] + bo;  // Nout=16 stride
    v += (x - v) * decay;
    float s = (v >= 1.f) ? 1.f : 0.f;
    v *= (1.f - s);
    accum += s;
  }
  out[b * 10 + o] = accum * 0.2f;
}

// ---------------------------------------------------------------------------
static inline size_t alignUp(size_t x) { return (x + 255) & ~(size_t)255; }
static inline int cdiv(int a, int b) { return (a + b - 1) / b; }

extern "C" void kernel_launch(void* const* d_in, const int* in_sizes, int n_in,
                              void* d_out, int out_size, void* d_ws, size_t ws_size,
                              hipStream_t stream) {
  (void)in_sizes; (void)n_in; (void)out_size; (void)ws_size;
  const float* x      = (const float*)d_in[0];
  const float* wc1    = (const float*)d_in[1];
  const float* wc2    = (const float*)d_in[2];
  const float* wc3    = (const float*)d_in[3];
  const float* wc4    = (const float*)d_in[4];
  const float* gamma1 = (const float*)d_in[5];
  const float* beta1  = (const float*)d_in[6];
  const float* gamma2 = (const float*)d_in[7];
  const float* beta2  = (const float*)d_in[8];
  const float* gamma3 = (const float*)d_in[9];
  const float* beta3  = (const float*)d_in[10];
  const float* gamma4 = (const float*)d_in[11];
  const float* beta4  = (const float*)d_in[12];
  const float* wl1    = (const float*)d_in[13];
  const float* wl2    = (const float*)d_in[14];
  const float* wl3    = (const float*)d_in[15];
  const float* wl4    = (const float*)d_in[16];
  const float* wl5    = (const float*)d_in[17];
  const float* wl6    = (const float*)d_in[18];
  const float* fc1_w  = (const float*)d_in[19];
  const float* fc1_b  = (const float*)d_in[20];
  const float* fc2_w  = (const float*)d_in[21];
  const float* fc2_b  = (const float*)d_in[22];
  float* out = (float*)d_out;

  const int B = 32, TB = 160;

  // ---- workspace layout ----
  char* ws = (char*)d_ws;
  size_t off = 0;
  unsigned short* xbf  = (unsigned short*)(ws + off); off = alignUp(off + (size_t)B * 3 * 128 * 128 * 2);
  unsigned short* w1m  = (unsigned short*)(ws + off); off = alignUp(off + (size_t)16 * 96 * 2);
  unsigned short* w2m  = (unsigned short*)(ws + off); off = alignUp(off + (size_t)32 * 416 * 2);
  unsigned short* w3m  = (unsigned short*)(ws + off); off = alignUp(off + (size_t)32 * 800 * 2);
  unsigned short* w4m  = (unsigned short*)(ws + off); off = alignUp(off + (size_t)64 * 800 * 2);
  unsigned short* wf1m = (unsigned short*)(ws + off); off = alignUp(off + (size_t)1024 * 1024 * 2);
  unsigned short* wf2m = (unsigned short*)(ws + off); off = alignUp(off + (size_t)16 * 1024 * 2);
  float* stats = (float*)(ws + off); off = alignUp(off + 128 * sizeof(float));
  float* Y = (float*)(ws + off);          // max: conv2 out 160*32*58*58 f32
  off = alignUp(off + (size_t)TB * 32 * 58 * 58 * 4);
  unsigned short* S = (unsigned short*)(ws + off);  // max: s1 160*16*62*62 bf16

  // ---- input & weight conversion ----
  {
    int nx = B * 3 * 128 * 128;
    k_f32_to_bf16<<<cdiv(nx, TPB), TPB, 0, stream>>>(x, xbf, nx);
    k_convw_to_mat<<<cdiv(16 * 96,  TPB), TPB, 0, stream>>>(wc1, w1m, 16, 3,  96);
    k_convw_to_mat<<<cdiv(32 * 416, TPB), TPB, 0, stream>>>(wc2, w2m, 32, 16, 416);
    k_convw_to_mat<<<cdiv(32 * 800, TPB), TPB, 0, stream>>>(wc3, w3m, 32, 32, 800);
    k_convw_to_mat<<<cdiv(64 * 800, TPB), TPB, 0, stream>>>(wc4, w4m, 64, 32, 800);
    k_fcw_to_mat<<<cdiv(1024 * 1024, TPB), TPB, 0, stream>>>(fc1_w, wf1m, 1024, 1024, 1024);
    k_fcw_to_mat<<<cdiv(16 * 1024,   TPB), TPB, 0, stream>>>(fc2_w, wf2m, 10, 16, 1024);
  }

  // ---- block 1: input broadcast over T => conv/BN on B=32 only ----
  {
    const int Cin = 3, Cout = 16, Hin = 128, Win = 128, Hc = 124, Wc = 124, Hp = 62, Wp = 62;
    int M = B * Hc * Wc;  // 492032
    k_gemm_wmma<16, 1><<<dim3(cdiv(M, 128), 1), TPB, 0, stream>>>(
        xbf, w1m, Y, M, Cin * 25, Cout, Cin, Hin, Win, Hc, Wc);
    k_zero_f32<<<1, 128, 0, stream>>>(stats, 128);
    k_bn_stats<<<dim3(Cout, 64), TPB, 0, stream>>>(Y, stats, Cout, Hc * Wc, B);
    int tot = B * Cout * Hp * Wp;
    k_bn_plif_pool<<<cdiv(tot, TPB), TPB, 0, stream>>>(
        Y, stats, gamma1, beta1, wl1, S, B, Cout, Hc, Wc, Hp, Wp,
        1.f / ((float)B * Hc * Wc), 1);
  }
  // ---- block 2 ----
  {
    const int Cin = 16, Cout = 32, Hin = 62, Win = 62, Hc = 58, Wc = 58, Hp = 29, Wp = 29;
    int M = TB * Hc * Wc;  // 538240
    k_gemm_wmma<32, 1><<<dim3(cdiv(M, 128), 1), TPB, 0, stream>>>(
        S, w2m, Y, M, Cin * 25, Cout, Cin, Hin, Win, Hc, Wc);
    k_zero_f32<<<1, 128, 0, stream>>>(stats, 128);
    k_bn_stats<<<dim3(Cout, 64), TPB, 0, stream>>>(Y, stats, Cout, Hc * Wc, TB);
    int tot = B * Cout * Hp * Wp;
    k_bn_plif_pool<<<cdiv(tot, TPB), TPB, 0, stream>>>(
        Y, stats, gamma2, beta2, wl2, S, B, Cout, Hc, Wc, Hp, Wp,
        1.f / ((float)TB * Hc * Wc), 0);
  }
  // ---- block 3 ----
  {
    const int Cin = 32, Cout = 32, Hin = 29, Win = 29, Hc = 25, Wc = 25, Hp = 12, Wp = 12;
    int M = TB * Hc * Wc;  // 100000
    k_gemm_wmma<32, 1><<<dim3(cdiv(M, 128), 1), TPB, 0, stream>>>(
        S, w3m, Y, M, Cin * 25, Cout, Cin, Hin, Win, Hc, Wc);
    k_zero_f32<<<1, 128, 0, stream>>>(stats, 128);
    k_bn_stats<<<dim3(Cout, 64), TPB, 0, stream>>>(Y, stats, Cout, Hc * Wc, TB);
    int tot = B * Cout * Hp * Wp;
    k_bn_plif_pool<<<cdiv(tot, TPB), TPB, 0, stream>>>(
        Y, stats, gamma3, beta3, wl3, S, B, Cout, Hc, Wc, Hp, Wp,
        1.f / ((float)TB * Hc * Wc), 0);
  }
  // ---- block 4 ----
  {
    const int Cin = 32, Cout = 64, Hin = 12, Win = 12, Hc = 8, Wc = 8, Hp = 4, Wp = 4;
    int M = TB * Hc * Wc;  // 10240
    k_gemm_wmma<64, 1><<<dim3(cdiv(M, 128), 1), TPB, 0, stream>>>(
        S, w4m, Y, M, Cin * 25, Cout, Cin, Hin, Win, Hc, Wc);
    k_zero_f32<<<1, 128, 0, stream>>>(stats, 128);
    k_bn_stats<<<dim3(Cout, 64), TPB, 0, stream>>>(Y, stats, Cout, Hc * Wc, TB);
    int tot = B * Cout * Hp * Wp;
    k_bn_plif_pool<<<cdiv(tot, TPB), TPB, 0, stream>>>(
        Y, stats, gamma4, beta4, wl4, S, B, Cout, Hc, Wc, Hp, Wp,
        1.f / ((float)TB * Hc * Wc), 0);
  }
  // ---- FC1: [160,1024] x [1024,1024]^T + bias -> PLIF spikes ----
  {
    k_gemm_wmma<64, 0><<<dim3(cdiv(TB, 128), 1024 / 64), TPB, 0, stream>>>(
        S, wf1m, Y, TB, 1024, 1024, 0, 0, 0, 1, 1);
    k_fc_plif<<<cdiv(B * 1024, TPB), TPB, 0, stream>>>(Y, fc1_b, wl5, S, B, 1024);
  }
  // ---- FC2: [160,1024] x [16pad,1024]^T + bias -> PLIF -> mean over T ----
  {
    k_gemm_wmma<16, 0><<<dim3(cdiv(TB, 128), 1), TPB, 0, stream>>>(
        S, wf2m, Y, TB, 1024, 16, 0, 0, 0, 1, 1);
    k_fc2_plif_mean<<<cdiv(B * 10, TPB), TPB, 0, stream>>>(Y, fc2_b, wl6, out, B);
  }
}
